// AlexNetOWT_BN_1537598292144
// MI455X (gfx1250) — 8-bit matrix operand order repaired and checked by reading, compiled, not run on hardware
//
#include <hip/hip_runtime.h>

typedef __attribute__((ext_vector_type(8))) int v8i;

static __device__ __forceinline__ float fsgn(float v){ return (float)((v > 0.f) - (v < 0.f)); }
static __device__ __forceinline__ char  fsgn8(float v){ return (char)((v > 0.f) - (v < 0.f)); }


__global__ void k0_prep(const float* __restrict__ w1, const float* __restrict__ w2,
                        const float* __restrict__ w3, const float* __restrict__ wfc,
                        unsigned long long* __restrict__ stats,
                        char* __restrict__ w1s, char* __restrict__ w2s,
                        char* __restrict__ w3s, char* __restrict__ wfcs) {
  const int t = threadIdx.x;
  for (int i = t; i < 6 * 48; i += 256) stats[i] = 0ull;
  for (int i = t; i < 48 * 192; i += 256) {
    const int n = i / 192, r = i - n * 192, b = r >> 4, ci = r & 15;
    w1s[i] = (b < 9 && ci < 12) ? fsgn8(w1[n * 108 + ci * 9 + b]) : (char)0;
  }
  for (int i = t; i < 48 * 576; i += 256) {
    const int n = i / 576, r = i - n * 576, b = r >> 6, ci = r & 63;
    w2s[i] = (ci < 48) ? fsgn8(w2[n * 432 + ci * 9 + b]) : (char)0;
  }
  for (int i = t; i < 48 * 576; i += 256) {
    const int n = i / 576, r = i - n * 576, b = r >> 6, ci = r & 63;
    w3s[i] = (ci < 48) ? fsgn8(w3[n * 432 + ci * 9 + b]) : (char)0;
  }
  for (int i = t; i < 5 * 2592; i += 256) wfcs[i] = fsgn8(wfc[i]);
}

__global__ void k_bn(const unsigned long long* __restrict__ sum,
                     const unsigned long long* __restrict__ sq,
                     const float* __restrict__ g, const float* __restrict__ b,
                     float* __restrict__ af, float* __restrict__ bf, float inv_count) {
  const int c = threadIdx.x;
  if (c < 48) {
    const float mean = (float)((double)(long long)sum[c] * (double)inv_count);
    const float msq  = (float)((double)(long long)sq[c]  * (double)inv_count);
    const float var  = msq - mean * mean;
    const float a = g[c] * rsqrtf(var + 1e-5f);
    af[c] = a;
    bf[c] = b[c] - mean * a;
  }
}

__global__ void __launch_bounds__(128) k1_conv1(const float* __restrict__ x,
        float* __restrict__ xb1, const char* __restrict__ w1s_g,
        char* __restrict__ pool1,
        unsigned long long* __restrict__ sum1, unsigned long long* __restrict__ sq1) {
  __shared__ __align__(16) unsigned char s_in[14 * 17 * 16];
  __shared__ __align__(16) int  s_w4[48 * 48];
  __shared__ __align__(16) char s_conv[48 * 192];
  __shared__ unsigned long long ch_s[48], ch_q[48];
  const int img = blockIdx.x, tid = threadIdx.x;

  for (int i = tid; i < 14 * 17 * 16 / 4; i += 128) ((int*)s_in)[i] = 0;
  for (int i = tid; i < 48 * 48 / 4; i += 128) ((int4*)s_w4)[i] = ((const int4*)w1s_g)[i];
  if (tid < 48) { ch_s[tid] = 0ull; ch_q[tid] = 0ull; }
  __syncthreads();

  const float* xi = x   + (size_t)img * 2160;
  float*       xo = xb1 + (size_t)img * 2160;
  for (int i = tid; i < 2160; i += 128) {
    const float s = fsgn(xi[i]);
    xo[i] = s;
    const int c = i / 180, p = i - c * 180, y = p / 15, xx = p - y * 15;
    s_in[((y + 1) * 17 + xx + 1) * 16 + c] = (unsigned char)(char)s;
  }
  __syncthreads();

  const int lane = tid & 31, wv = tid >> 5;
  const int half = lane >> 4, lm = lane & 15;
  const int half8 = half << 3;
  for (int mt = wv; mt < 12; mt += 4) {
    const int m  = mt * 16 + lm;
    const int mc = (m < 180) ? m : 179;
    const int my = mc / 15, mx = mc - my * 15;
    const int ab = (my * 17 + mx) * 16 + half8;
    v8i acc[3];
    #pragma unroll
    for (int nt = 0; nt < 3; nt++) acc[nt] = (v8i){0,0,0,0,0,0,0,0};
    #pragma unroll
    for (int kc = 0; kc < 3; kc++) {
      v8i A;
      #pragma unroll
      for (int p = 0; p < 4; p++) {
        const int blk = kc * 4 + p;
        if (blk < 9) {
          const int ky = blk / 3, kx = blk - ky * 3;
          const uint2 a = *(const uint2*)&s_in[ab + (ky * 17 + kx) * 16];
          A[2 * p] = (int)a.x; A[2 * p + 1] = (int)a.y;
        } else { A[2 * p] = 0; A[2 * p + 1] = 0; }
      }
      #pragma unroll
      for (int nt = 0; nt < 3; nt++) {
        const int bi = (nt * 16 + lm) * 48 + kc * 16 + half * 2;
        const int2 b0 = *(const int2*)&s_w4[bi];
        const int2 b1 = *(const int2*)&s_w4[bi + 4];
        const int2 b2 = *(const int2*)&s_w4[bi + 8];
        const int2 b3 = *(const int2*)&s_w4[bi + 12];
        v8i Bv;
        Bv[0] = b0.x; Bv[1] = b0.y; Bv[2] = b1.x; Bv[3] = b1.y;
        Bv[4] = b2.x; Bv[5] = b2.y; Bv[6] = b3.x; Bv[7] = b3.y;
        acc[nt] = __builtin_amdgcn_wmma_i32_16x16x64_iu8(true, A, true, Bv, acc[nt], false, false);
      }
      asm volatile("" ::: "memory");
    }
    #pragma unroll
    for (int nt = 0; nt < 3; nt++) {
      const int n = nt * 16 + lm;
      #pragma unroll
      for (int r = 0; r < 8; r++) {
        const int mm = mt * 16 + r + half * 8;
        s_conv[n * 192 + mm] = (char)acc[nt][r];
      }
    }
  }
  __syncthreads();

  char* po = pool1 + (size_t)img * (48 * 154);
  for (int i = tid; i < 48 * 154; i += 128) {
    const int c = i / 154, r = i - c * 154, py = r / 14, px = r - py * 14;
    const char* bse = &s_conv[c * 192 + py * 15 + px];
    int p = bse[0], q = bse[1];
    if (q > p) p = q;  q = bse[15]; if (q > p) p = q;  q = bse[16]; if (q > p) p = q;
    po[i] = (char)p;
    atomicAdd(&ch_s[c], (unsigned long long)(long long)p);
    atomicAdd(&ch_q[c], (unsigned long long)(long long)(p * p));
  }
  __syncthreads();
  if (tid < 48) {
    atomicAdd(&sum1[tid], ch_s[tid]);
    atomicAdd(&sq1[tid],  ch_q[tid]);
  }
}

__global__ void __launch_bounds__(128) k3_conv2(const char* __restrict__ pool1,
        const float* __restrict__ af, const float* __restrict__ bf,
        float* __restrict__ xb2, const char* __restrict__ w2s_g,
        short* __restrict__ c2,
        unsigned long long* __restrict__ sum2, unsigned long long* __restrict__ sq2) {
  __shared__ __align__(16) unsigned char s_a[154 * 48];
  __shared__ __align__(16) int   s_w4[48 * 144];
  __shared__ __align__(16) short s_c2[48 * 112];
  __shared__ unsigned long long ch_s[48], ch_q[48];
  __shared__ float s_af[48], s_bf[48];
  const int img = blockIdx.x, tid = threadIdx.x;

  if (tid < 48) { s_af[tid] = af[tid]; s_bf[tid] = bf[tid]; ch_s[tid] = 0ull; ch_q[tid] = 0ull; }
  for (int i = tid; i < 48 * 144 / 4; i += 128) ((int4*)s_w4)[i] = ((const int4*)w2s_g)[i];
  __syncthreads();

  const char* pin = pool1 + (size_t)img * (48 * 154);
  float*      xo  = xb2   + (size_t)img * (48 * 154);
  for (int i = tid; i < 48 * 154; i += 128) {
    const int c = i / 154, p = i - c * 154;
    const float s = fsgn(s_af[c] * (float)pin[i] + s_bf[c]);
    xo[i] = s;
    s_a[p * 48 + c] = (unsigned char)(char)s;
  }
  __syncthreads();

  const int lane = tid & 31, wv = tid >> 5;
  const int half = lane >> 4, lm = lane & 15;
  const int half8 = half << 3;
  short* co16 = c2 + (size_t)img * (48 * 108);
  for (int mt = wv; mt < 7; mt += 4) {
    const int m  = mt * 16 + lm;
    const int mc = (m < 108) ? m : 107;
    const int my = mc / 12, mx = mc - my * 12;
    const int ab = (my * 14 + mx) * 48 + half8;
    v8i acc[3];
    #pragma unroll
    for (int nt = 0; nt < 3; nt++) acc[nt] = (v8i){0,0,0,0,0,0,0,0};
    #pragma unroll
    for (int kc = 0; kc < 9; kc++) {
      const int ky = kc / 3, kx = kc - ky * 3;
      const int boff = (ky * 14 + kx) * 48;
      v8i A;
      {
        const uint2 a01 = *(const uint2*)&s_a[ab + boff];
        const uint2 a23 = *(const uint2*)&s_a[ab + boff + 16];
        const uint2 a45 = *(const uint2*)&s_a[ab + boff + 32];
        A[0] = (int)a01.x; A[1] = (int)a01.y;
        A[2] = (int)a23.x; A[3] = (int)a23.y;
        A[4] = (int)a45.x; A[5] = (int)a45.y;
        A[6] = 0; A[7] = 0;
      }
      #pragma unroll
      for (int nt = 0; nt < 3; nt++) {
        const int bi = (nt * 16 + lm) * 144 + kc * 16 + half * 2;
        const int2 b0 = *(const int2*)&s_w4[bi];
        const int2 b1 = *(const int2*)&s_w4[bi + 4];
        const int2 b2 = *(const int2*)&s_w4[bi + 8];
        const int2 b3 = *(const int2*)&s_w4[bi + 12];
        v8i Bv;
        Bv[0] = b0.x; Bv[1] = b0.y; Bv[2] = b1.x; Bv[3] = b1.y;
        Bv[4] = b2.x; Bv[5] = b2.y; Bv[6] = b3.x; Bv[7] = b3.y;
        acc[nt] = __builtin_amdgcn_wmma_i32_16x16x64_iu8(true, A, true, Bv, acc[nt], false, false);
      }
      asm volatile("" ::: "memory");
    }
    #pragma unroll
    for (int nt = 0; nt < 3; nt++) {
      const int n = nt * 16 + lm;
      #pragma unroll
      for (int r = 0; r < 8; r++) {
        const int mm = mt * 16 + r + half * 8;
        s_c2[n * 112 + mm] = (short)acc[nt][r];
      }
    }
  }
  __syncthreads();

  int* co = (int*)co16;
  for (int i = tid; i < 48 * 54; i += 128) {
    const int c = i / 54, r = i - c * 54;
    const int w = ((const int*)s_c2)[c * 56 + r];
    co[i] = w;
    const short s0 = (short)(w & 0xffff);
    const short s1 = (short)(w >> 16);
    atomicAdd(&ch_s[c], (unsigned long long)(long long)((int)s0 + (int)s1));
    atomicAdd(&ch_q[c], (unsigned long long)(long long)((int)s0 * s0 + (int)s1 * s1));
  }
  __syncthreads();
  if (tid < 48) {
    atomicAdd(&sum2[tid], ch_s[tid]);
    atomicAdd(&sq2[tid],  ch_q[tid]);
  }
}

__global__ void __launch_bounds__(128) k5_conv3(const short* __restrict__ c2,
        const float* __restrict__ af, const float* __restrict__ bf,
        float* __restrict__ xb3, const char* __restrict__ w3s_g,
        short* __restrict__ pool3,
        unsigned long long* __restrict__ sum3, unsigned long long* __restrict__ sq3) {
  __shared__ __align__(16) unsigned char s_a[108 * 48];
  __shared__ __align__(16) int   s_w4[48 * 144];
  __shared__ __align__(16) short s_c3[48 * 80];
  __shared__ unsigned long long ch_s[48], ch_q[48];
  __shared__ float s_af[48], s_bf[48];
  const int img = blockIdx.x, tid = threadIdx.x;

  if (tid < 48) { s_af[tid] = af[tid]; s_bf[tid] = bf[tid]; ch_s[tid] = 0ull; ch_q[tid] = 0ull; }
  for (int i = tid; i < 48 * 144 / 4; i += 128) ((int4*)s_w4)[i] = ((const int4*)w3s_g)[i];
  __syncthreads();

  const short* pin = c2  + (size_t)img * (48 * 108);
  float*       xo  = xb3 + (size_t)img * (48 * 108);
  for (int i = tid; i < 48 * 108; i += 128) {
    const int c = i / 108, p = i - c * 108;
    const float s = fsgn(s_af[c] * (float)pin[i] + s_bf[c]);
    xo[i] = s;
    s_a[p * 48 + c] = (unsigned char)(char)s;
  }
  __syncthreads();

  const int lane = tid & 31, wv = tid >> 5;
  const int half = lane >> 4, lm = lane & 15;
  const int half8 = half << 3;
  for (int mt = wv; mt < 5; mt += 4) {
    const int m  = mt * 16 + lm;
    const int mc = (m < 70) ? m : 69;
    const int my = mc / 10, mx = mc - my * 10;
    const int ab = (my * 12 + mx) * 48 + half8;
    v8i acc[3];
    #pragma unroll
    for (int nt = 0; nt < 3; nt++) acc[nt] = (v8i){0,0,0,0,0,0,0,0};
    #pragma unroll
    for (int kc = 0; kc < 9; kc++) {
      const int ky = kc / 3, kx = kc - ky * 3;
      const int boff = (ky * 12 + kx) * 48;
      v8i A;
      {
        const uint2 a01 = *(const uint2*)&s_a[ab + boff];
        const uint2 a23 = *(const uint2*)&s_a[ab + boff + 16];
        const uint2 a45 = *(const uint2*)&s_a[ab + boff + 32];
        A[0] = (int)a01.x; A[1] = (int)a01.y;
        A[2] = (int)a23.x; A[3] = (int)a23.y;
        A[4] = (int)a45.x; A[5] = (int)a45.y;
        A[6] = 0; A[7] = 0;
      }
      #pragma unroll
      for (int nt = 0; nt < 3; nt++) {
        const int bi = (nt * 16 + lm) * 144 + kc * 16 + half * 2;
        const int2 b0 = *(const int2*)&s_w4[bi];
        const int2 b1 = *(const int2*)&s_w4[bi + 4];
        const int2 b2 = *(const int2*)&s_w4[bi + 8];
        const int2 b3 = *(const int2*)&s_w4[bi + 12];
        v8i Bv;
        Bv[0] = b0.x; Bv[1] = b0.y; Bv[2] = b1.x; Bv[3] = b1.y;
        Bv[4] = b2.x; Bv[5] = b2.y; Bv[6] = b3.x; Bv[7] = b3.y;
        acc[nt] = __builtin_amdgcn_wmma_i32_16x16x64_iu8(true, A, true, Bv, acc[nt], false, false);
      }
      asm volatile("" ::: "memory");
    }
    #pragma unroll
    for (int nt = 0; nt < 3; nt++) {
      const int n = nt * 16 + lm;
      #pragma unroll
      for (int r = 0; r < 8; r++) {
        const int mm = mt * 16 + r + half * 8;
        s_c3[n * 80 + mm] = (short)acc[nt][r];
      }
    }
  }
  __syncthreads();

  short* po = pool3 + (size_t)img * (48 * 54);
  for (int i = tid; i < 48 * 54; i += 128) {
    const int c = i / 54, r = i - c * 54, py = r / 9, px = r - py * 9;
    const short* bse = &s_c3[c * 80 + py * 10 + px];
    int p = bse[0], q = bse[1];
    if (q > p) p = q;  q = bse[10]; if (q > p) p = q;  q = bse[11]; if (q > p) p = q;
    po[i] = (short)p;
    atomicAdd(&ch_s[c], (unsigned long long)(long long)p);
    atomicAdd(&ch_q[c], (unsigned long long)(long long)(p * p));
  }
  __syncthreads();
  if (tid < 48) {
    atomicAdd(&sum3[tid], ch_s[tid]);
    atomicAdd(&sq3[tid],  ch_q[tid]);
  }
}

__global__ void __launch_bounds__(128) k7_fc(const short* __restrict__ pool3,
        const float* __restrict__ af, const float* __restrict__ bf,
        const char* __restrict__ wfcs, float* __restrict__ fb, float* __restrict__ out) {
  __shared__ char  s_f[2592];
  __shared__ int   s_red[5];
  __shared__ float s_af[48], s_bf[48];
  const int img = blockIdx.x, tid = threadIdx.x;

  if (tid < 48) { s_af[tid] = af[tid]; s_bf[tid] = bf[tid]; }
  if (tid < 5) s_red[tid] = 0;
  __syncthreads();

  const short* pin = pool3 + (size_t)img * 2592;
  float*       fo  = fb    + (size_t)img * 2592;
  for (int i = tid; i < 2592; i += 128) {
    const int c = i / 54;
    const float s = fsgn(s_af[c] * (float)pin[i] + s_bf[c]);
    fo[i] = s;
    s_f[i] = (char)s;
  }
  __syncthreads();

  int p0 = 0, p1 = 0, p2 = 0, p3 = 0, p4 = 0;
  for (int i = tid; i < 2592; i += 128) {
    const int fv = (int)s_f[i];
    p0 += fv * (int)wfcs[i];
    p1 += fv * (int)wfcs[2592 + i];
    p2 += fv * (int)wfcs[2 * 2592 + i];
    p3 += fv * (int)wfcs[3 * 2592 + i];
    p4 += fv * (int)wfcs[4 * 2592 + i];
  }
  atomicAdd(&s_red[0], p0);
  atomicAdd(&s_red[1], p1);
  atomicAdd(&s_red[2], p2);
  atomicAdd(&s_red[3], p3);
  atomicAdd(&s_red[4], p4);
  __syncthreads();
  if (tid < 5) out[(size_t)img * 5 + tid] = (float)s_red[tid];
}

extern "C" void kernel_launch(void* const* d_in, const int* in_sizes, int n_in,
                              void* d_out, int out_size, void* d_ws, size_t ws_size,
                              hipStream_t stream) {
  (void)in_sizes; (void)n_in; (void)out_size; (void)ws_size;
  const float* x   = (const float*)d_in[0];
  const float* w1  = (const float*)d_in[1];
  const float* g1  = (const float*)d_in[2];
  const float* b1  = (const float*)d_in[3];
  const float* w2  = (const float*)d_in[4];
  const float* g2  = (const float*)d_in[5];
  const float* b2  = (const float*)d_in[6];
  const float* w3  = (const float*)d_in[7];
  const float* g3  = (const float*)d_in[8];
  const float* b3  = (const float*)d_in[9];
  const float* wfc = (const float*)d_in[10];

  char* ws = (char*)d_ws;
  unsigned long long* stats = (unsigned long long*)(ws + 0);
  unsigned long long* sum1 = stats +   0, * sq1 = stats +  48;
  unsigned long long* sum2 = stats +  96, * sq2 = stats + 144;
  unsigned long long* sum3 = stats + 192, * sq3 = stats + 240;
  float* af1 = (float*)(ws + 2304), * bf1 = (float*)(ws + 2496);
  float* af2 = (float*)(ws + 2688), * bf2 = (float*)(ws + 2880);
  float* af3 = (float*)(ws + 3072), * bf3 = (float*)(ws + 3264);
  char*  w1s  = ws + 3584;
  char*  w2s  = ws + 12800;
  char*  w3s  = ws + 40448;
  char*  wfcs = ws + 68096;
  char*  pool1 = ws + 81152;
  short* c2    = (short*)(ws + 30358784ull);
  short* pool3 = (short*)(ws + 72826112ull);

  float* o     = (float*)d_out;
  float* o_out = o;
  float* o_xb1 = o + 20480ull;
  float* o_xb2 = o + 8867840ull;
  float* o_xb3 = o + 39145472ull;
  float* o_fb  = o + 60379136ull;

  k0_prep<<<1, 256, 0, stream>>>(w1, w2, w3, wfc, stats, w1s, w2s, w3s, wfcs);
  k1_conv1<<<4096, 128, 0, stream>>>(x, o_xb1, w1s, pool1, sum1, sq1);
  k_bn<<<1, 64, 0, stream>>>(sum1, sq1, g1, b1, af1, bf1, 1.0f / 630784.0f);
  k3_conv2<<<4096, 128, 0, stream>>>(pool1, af1, bf1, o_xb2, w2s, c2, sum2, sq2);
  k_bn<<<1, 64, 0, stream>>>(sum2, sq2, g2, b2, af2, bf2, 1.0f / 442368.0f);
  k5_conv3<<<4096, 128, 0, stream>>>(c2, af2, bf2, o_xb3, w3s, pool3, sum3, sq3);
  k_bn<<<1, 64, 0, stream>>>(sum3, sq3, g3, b3, af3, bf3, 1.0f / 221184.0f);
  k7_fc<<<4096, 128, 0, stream>>>(pool3, af3, bf3, wfcs, o_fb, o_out);
}
